// LorentzTransformer_49770081026292
// MI455X (gfx1250) — compile-verified, matrix operand order checked by reading, not run on hardware
//
#include <hip/hip_runtime.h>
#include <math.h>

// ---------------- model constants ----------------
#define CB    2
#define CL    1024
#define CD    768
#define CH    12
#define CDH   64
#define CDFF  3072
#define CNL   4
#define CVOCAB 50257

// ---------------- WMMA GEMM tiling ----------------
#define BM 128
#define BN 128
#define BK 32
#define KPADB 56   // bf16 elems per LDS row: 112B -> 16B-aligned chunks, staggered banks

typedef __attribute__((ext_vector_type(16))) __bf16 v16bf;
typedef __attribute__((ext_vector_type(8)))  float  v8f;

// ---- gfx1250 async global->LDS copy (ASYNCcnt-tracked), ISA §10.7/15.18 ----
__device__ __forceinline__ void async_copy16(const void* gsrc, void* ldst) {
#if defined(__gfx1250__)
  unsigned int loff = (unsigned int)(unsigned long long)(uintptr_t)ldst;
  asm volatile("global_load_async_to_lds_b128 %0, %1, off"
               :: "v"(loff), "v"(gsrc) : "memory");
#else
  *(uint4*)ldst = *(const uint4*)gsrc;
#endif
}

__device__ __forceinline__ void async_wait0() {
#if defined(__gfx1250__)
#if __has_builtin(__builtin_amdgcn_s_wait_asynccnt)
  __builtin_amdgcn_s_wait_asynccnt(0);
#else
  asm volatile("s_wait_asynccnt 0x0" ::: "memory");
#endif
#endif
}

// Fragment per CDNA5 16-bit A-layout: lanes0-15 = rows, lane>=16 shifts K by 8;
// elems 0..7 -> K=ks..ks+7, elems 8..15 -> K=ks+16..ks+23 (two aligned 16B chunks).
__device__ __forceinline__ v16bf frag_from_lds(const __bf16* base, int rowBase, int lane) {
  const __bf16* p = base + (rowBase + (lane & 15)) * KPADB + ((lane >> 4) << 3);
  v16bf f;
#pragma unroll
  for (int j = 0; j < 8; ++j) f[j] = p[j];
#pragma unroll
  for (int j = 0; j < 8; ++j) f[8 + j] = p[16 + j];
  return f;
}

// Issue async fills for one 128x32 bf16 tile. 512 16B-chunks, 2 per thread.
// OOB rows clamp to the last valid row: their products only land in C rows/cols
// that the epilogue never stores.
__device__ __forceinline__ void fill_tile(__bf16* lds, const __bf16* g,
                                          int start, int lastRow, int ld,
                                          int k0, int tid) {
#pragma unroll
  for (int c = 0; c < 2; ++c) {
    const int chunk = tid + (c << 8);
    const int row = chunk >> 2;
    const int cc  = (chunk & 3) << 3;
    int gr = start + row;
    if (gr > lastRow) gr = lastRow;
    async_copy16(g + (long long)gr * ld + k0 + cc, lds + row * KPADB + cc);
  }
}

// Generic NT GEMM: C[M,N] = epi( A[M,K] * B[N,K]^T ), A/B bf16 in memory,
// fp32 accumulate, C fp32 or bf16. Double-buffered async-LDS pipeline.
__global__ __launch_bounds__(256) void wmma_gemm_kernel(
    const __bf16* A, const __bf16* B, void* Cv,
    const float* bias, const float* resid, const float* cscale,
    int M, int N, int K, int lda, int ldb, int ldc,
    int gelu, int causal, int outBf16, int zdiv,
    long long aOffB, long long aOffH, long long bOffB, long long bOffH,
    long long cOffB, long long cOffH)
{
  const int z  = blockIdx.z;
  const int bz = z / zdiv, hz = z % zdiv;
  A += (long long)bz * aOffB + (long long)hz * aOffH;
  B += (long long)bz * bOffB + (long long)hz * bOffH;
  const long long coff = (long long)bz * cOffB + (long long)hz * cOffH;
  if (resid) resid += coff;

  const int mBlock = blockIdx.y * BM;
  const int nBlock = blockIdx.x * BN;
  if (causal && nBlock > mBlock + (BM - 1)) return;   // fully-masked tile

  __shared__ __align__(16) __bf16 lA[2][BM * KPADB];
  __shared__ __align__(16) __bf16 lB[2][BN * KPADB];

  const int tid  = threadIdx.x;
  const int lane = tid & 31;
  const int wave = tid >> 5;          // 8 wave32 waves
  const int wm   = (wave >> 1) * 32;  // 0,32,64,96
  const int wn   = (wave & 1) * 64;   // 0,64

  v8f acc[2][4];
#pragma unroll
  for (int i = 0; i < 2; ++i)
#pragma unroll
    for (int j = 0; j < 4; ++j)
#pragma unroll
      for (int e = 0; e < 8; ++e) acc[i][j][e] = 0.f;

  const int nk = K / BK;
  fill_tile(lA[0], A, mBlock, M - 1, lda, 0, tid);
  fill_tile(lB[0], B, nBlock, N - 1, ldb, 0, tid);

  for (int kt = 0; kt < nk; ++kt) {
    const int cur = kt & 1;
    async_wait0();
    __syncthreads();                        // tile kt visible to all waves
    if (kt + 1 < nk) {                      // prefetch next tile into other buffer
      fill_tile(lA[cur ^ 1], A, mBlock, M - 1, lda, (kt + 1) * BK, tid);
      fill_tile(lB[cur ^ 1], B, nBlock, N - 1, ldb, (kt + 1) * BK, tid);
    }
    if (kt + 2 < nk) {                      // gfx1250 global_prefetch for k+2
      __builtin_prefetch(A + (long long)(mBlock + (tid >> 2)) * lda + (kt + 2) * BK, 0, 1);
      __builtin_prefetch(B + (long long)(nBlock + (tid >> 2)) * ldb + (kt + 2) * BK, 0, 1);
    }

    v16bf af[2], bfr[4];
    af[0] = frag_from_lds(lA[cur], wm, lane);
    af[1] = frag_from_lds(lA[cur], wm + 16, lane);
#pragma unroll
    for (int j = 0; j < 4; ++j) bfr[j] = frag_from_lds(lB[cur], wn + j * 16, lane);

#pragma unroll
    for (int i = 0; i < 2; ++i)
#pragma unroll
      for (int j = 0; j < 4; ++j)
        acc[i][j] = __builtin_amdgcn_wmma_f32_16x16x32_bf16(
            false, af[i], false, bfr[j], (short)0, acc[i][j], false, false);
  }

  // epilogue; C/D layout: VGPR e, lanes 0-15 -> M=e, lanes 16-31 -> M=e+8
  float*  Cf = (float*)Cv + coff;
  __bf16* Cb = (__bf16*)Cv + coff;
  const int colLane = lane & 15;
  const int rhalf   = (lane >> 4) << 3;
#pragma unroll
  for (int i = 0; i < 2; ++i) {
#pragma unroll
    for (int j = 0; j < 4; ++j) {
      const int col = nBlock + wn + j * 16 + colLane;
      if (col >= N) continue;
      const float cs = cscale ? cscale[col] : 1.f;
      const float bb = bias ? bias[col] : 0.f;
#pragma unroll
      for (int e = 0; e < 8; ++e) {
        const int row = mBlock + wm + i * 16 + rhalf + e;
        if (row >= M) continue;
        float val = acc[i][j][e] * cs + bb;
        if (gelu) val = 0.5f * val * (1.f + erff(val * 0.70710678118654752f));
        const long long idx = (long long)row * ldc + col;
        if (resid) val += resid[idx];
        if (outBf16) Cb[idx] = (__bf16)val; else Cf[idx] = val;
      }
    }
  }
}

// ---------------- support kernels ----------------

__global__ void cvt_bf16_kernel(const float* x, __bf16* y, int n4) {
  int i = blockIdx.x * blockDim.x + threadIdx.x;
  if (i >= n4) return;
  float4 f = ((const float4*)x)[i];
  __bf16* d = y + (long long)i * 4;
  d[0] = (__bf16)f.x; d[1] = (__bf16)f.y; d[2] = (__bf16)f.z; d[3] = (__bf16)f.w;
}

// fac[layer][d] = (1 - 2*alpha*mask) / sqrt(DH) = (1 - 0.5*m) * 0.125
__global__ void fac_kernel(const int* mask, float* fac, int n) {
  int i = blockIdx.x * blockDim.x + threadIdx.x;
  if (i >= n) return;
  fac[i] = (1.f - 0.5f * (float)mask[i]) * 0.125f;
}

__global__ void embed_kernel(const int* tokens, const float* embed,
                             const float* pos, float* h, int total) {
  int i = blockIdx.x * blockDim.x + threadIdx.x;
  if (i >= total) return;
  int d = i % CD;
  int row = i / CD;
  int l = row % CL;
  h[i] = embed[(long long)tokens[row] * CD + d] + pos[l * CD + d];
}

__global__ __launch_bounds__(256) void layernorm_kernel(
    const float* x, const float* g, const float* b, __bf16* y, int D) {
  __shared__ float red[256];
  const int row = blockIdx.x;
  const float* xr = x + (long long)row * D;
  __bf16* yr = y + (long long)row * D;
  const int tid = threadIdx.x;

  float s = 0.f;
  for (int d = tid; d < D; d += 256) s += xr[d];
  red[tid] = s; __syncthreads();
  for (int off = 128; off > 0; off >>= 1) {
    if (tid < off) red[tid] += red[tid + off];
    __syncthreads();
  }
  const float mu = red[0] / (float)D;
  __syncthreads();

  float v = 0.f;
  for (int d = tid; d < D; d += 256) { float t = xr[d] - mu; v += t * t; }
  red[tid] = v; __syncthreads();
  for (int off = 128; off > 0; off >>= 1) {
    if (tid < off) red[tid] += red[tid + off];
    __syncthreads();
  }
  const float rstd = rsqrtf(red[0] / (float)D + 1e-5f);

  for (int d = tid; d < D; d += 256)
    yr[d] = (__bf16)((xr[d] - mu) * rstd * g[d] + b[d]);
}

// vt[b][h][d][l] = v[b][l][h*DH+d]   (bf16)
__global__ void transpose_v_kernel(const __bf16* v, __bf16* vt, int total) {
  int i = blockIdx.x * blockDim.x + threadIdx.x;
  if (i >= total) return;
  int l = i % CL; int rest = i / CL;
  int d = rest % CDH; rest /= CDH;
  int h = rest % CH; int b = rest / CH;
  vt[i] = v[((long long)(b * CL + l)) * CD + h * CDH + d];
}

// in-place causal softmax over bf16 rows of (B*H, L, L)
__global__ __launch_bounds__(256) void softmax_kernel(__bf16* S, int L) {
  __shared__ float red[256];
  const int rowG = blockIdx.x;      // (b*H+h)*L + r
  const int r = rowG % L;
  __bf16* row = S + (long long)rowG * L;
  const int tid = threadIdx.x;

  float mx = -3.0e38f;
  for (int c = tid; c <= r; c += 256) mx = fmaxf(mx, (float)row[c]);
  red[tid] = mx; __syncthreads();
  for (int off = 128; off > 0; off >>= 1) {
    if (tid < off) red[tid] = fmaxf(red[tid], red[tid + off]);
    __syncthreads();
  }
  mx = red[0]; __syncthreads();

  float s = 0.f;
  for (int c = tid; c <= r; c += 256) s += expf((float)row[c] - mx);
  red[tid] = s; __syncthreads();
  for (int off = 128; off > 0; off >>= 1) {
    if (tid < off) red[tid] += red[tid + off];
    __syncthreads();
  }
  const float inv = 1.f / red[0];

  for (int c = tid; c < L; c += 256)
    row[c] = (c <= r) ? (__bf16)(expf((float)row[c] - mx) * inv) : (__bf16)0.f;
}

// ---------------- host side ----------------

static inline void gemm(hipStream_t s, const __bf16* A, const __bf16* B, void* C,
                        const float* bias, const float* resid, const float* cscale,
                        int M, int N, int K, int lda, int ldb, int ldc,
                        int gelu, int causal, int outBf16,
                        int Z, int zdiv,
                        long long aOffB, long long aOffH,
                        long long bOffB, long long bOffH,
                        long long cOffB, long long cOffH) {
  dim3 grid((N + BN - 1) / BN, (M + BM - 1) / BM, Z);
  wmma_gemm_kernel<<<grid, dim3(256), 0, s>>>(A, B, C, bias, resid, cscale,
      M, N, K, lda, ldb, ldc, gelu, causal, outBf16, zdiv,
      aOffB, aOffH, bOffB, bOffH, cOffB, cOffH);
}

static inline void cvt(hipStream_t s, const float* x, __bf16* y, long long n) {
  int n4 = (int)(n / 4);
  cvt_bf16_kernel<<<(n4 + 255) / 256, 256, 0, s>>>(x, y, n4);
}

extern "C" void kernel_launch(void* const* d_in, const int* in_sizes, int n_in,
                              void* d_out, int out_size, void* d_ws, size_t ws_size,
                              hipStream_t stream) {
  const int*   tokens  = (const int*)  d_in[0];
  const int*   tmask   = (const int*)  d_in[1];
  const float* embed   = (const float*)d_in[2];
  const float* pos_emb = (const float*)d_in[3];
  const float* wq      = (const float*)d_in[4];
  const float* wk      = (const float*)d_in[5];
  const float* wv      = (const float*)d_in[6];
  const float* wo      = (const float*)d_in[7];
  const float* ln1_g   = (const float*)d_in[8];
  const float* ln1_b   = (const float*)d_in[9];
  const float* ln2_g   = (const float*)d_in[10];
  const float* ln2_b   = (const float*)d_in[11];
  const float* ff_w1   = (const float*)d_in[12];
  const float* ff_b1   = (const float*)d_in[13];
  const float* ff_w2   = (const float*)d_in[14];
  const float* ff_b2   = (const float*)d_in[15];
  const float* lnf_g   = (const float*)d_in[16];
  const float* lnf_b   = (const float*)d_in[17];
  float* logits = (float*)d_out;
  (void)in_sizes; (void)n_in; (void)out_size; (void)ws_size;

  const int ROWS = CB * CL;   // 2048
  char* base = (char*)d_ws;
  auto alloc = [&](size_t bytes) -> char* {
    char* r = base; base += (bytes + 255) & ~(size_t)255; return r;
  };
  float*  h     = (float*) alloc((size_t)ROWS * CD * 4);
  __bf16* xnb   = (__bf16*)alloc((size_t)ROWS * CD * 2);
  __bf16* qb    = (__bf16*)alloc((size_t)ROWS * CD * 2);
  __bf16* kb    = (__bf16*)alloc((size_t)ROWS * CD * 2);
  __bf16* vb    = (__bf16*)alloc((size_t)ROWS * CD * 2);
  __bf16* vtb   = (__bf16*)alloc((size_t)ROWS * CD * 2);   // (B,H,DH,L)
  __bf16* attnb = (__bf16*)alloc((size_t)ROWS * CD * 2);
  __bf16* ffab  = (__bf16*)alloc((size_t)ROWS * CDFF * 2);
  __bf16* scb   = (__bf16*)alloc((size_t)CB * CH * CL * CL * 2);
  __bf16* wqb   = (__bf16*)alloc((size_t)CNL * CD * CD * 2);
  __bf16* wkb   = (__bf16*)alloc((size_t)CNL * CD * CD * 2);
  __bf16* wvb   = (__bf16*)alloc((size_t)CNL * CD * CD * 2);
  __bf16* wob   = (__bf16*)alloc((size_t)CNL * CD * CD * 2);
  __bf16* w1b   = (__bf16*)alloc((size_t)CNL * CDFF * CD * 2);
  __bf16* w2b   = (__bf16*)alloc((size_t)CNL * CD * CDFF * 2);
  __bf16* embb  = (__bf16*)alloc((size_t)CVOCAB * CD * 2);
  float*  fac   = (float*) alloc((size_t)CNL * CD * 4);

  // one-time (per launch) bf16 conversions of all GEMM operands
  cvt(stream, wq, wqb, (long long)CNL * CD * CD);
  cvt(stream, wk, wkb, (long long)CNL * CD * CD);
  cvt(stream, wv, wvb, (long long)CNL * CD * CD);
  cvt(stream, wo, wob, (long long)CNL * CD * CD);
  cvt(stream, ff_w1, w1b, (long long)CNL * CDFF * CD);
  cvt(stream, ff_w2, w2b, (long long)CNL * CD * CDFF);
  cvt(stream, embed, embb, (long long)CVOCAB * CD);
  fac_kernel<<<(CNL * CD + 255) / 256, 256, 0, stream>>>(tmask, fac, CNL * CD);

  const int totHD = ROWS * CD;
  embed_kernel<<<(totHD + 255) / 256, 256, 0, stream>>>(tokens, embed, pos_emb, h, totHD);

  for (int i = 0; i < CNL; ++i) {
    const __bf16* Wq = wqb + (size_t)i * CD * CD;
    const __bf16* Wk = wkb + (size_t)i * CD * CD;
    const __bf16* Wv = wvb + (size_t)i * CD * CD;
    const __bf16* Wo = wob + (size_t)i * CD * CD;
    const __bf16* W1 = w1b + (size_t)i * CDFF * CD;
    const __bf16* W2 = w2b + (size_t)i * CD * CDFF;

    layernorm_kernel<<<ROWS, 256, 0, stream>>>(h, ln1_g + i * CD, ln1_b + i * CD, xnb, CD);

    // Q (with folded Minkowski mask + 1/sqrt(DH) column scale), K, V
    gemm(stream, xnb, Wq, qb, nullptr, nullptr, fac + i * CD,
         ROWS, CD, CD, CD, CD, CD, 0, 0, 1, 1, 1, 0,0,0,0,0,0);
    gemm(stream, xnb, Wk, kb, nullptr, nullptr, nullptr,
         ROWS, CD, CD, CD, CD, CD, 0, 0, 1, 1, 1, 0,0,0,0,0,0);
    gemm(stream, xnb, Wv, vb, nullptr, nullptr, nullptr,
         ROWS, CD, CD, CD, CD, CD, 0, 0, 1, 1, 1, 0,0,0,0,0,0);

    transpose_v_kernel<<<(totHD + 255) / 256, 256, 0, stream>>>(vb, vtb, totHD);

    // scores = Qeff @ K^T per (b,h): M=N=L, K=DH, causal-skipped tiles
    gemm(stream, qb, kb, scb, nullptr, nullptr, nullptr,
         CL, CL, CDH, CD, CD, CL, 0, 1, 1,
         CB * CH, CH,
         (long long)CL * CD, CDH,
         (long long)CL * CD, CDH,
         (long long)CH * CL * CL, (long long)CL * CL);

    softmax_kernel<<<CB * CH * CL, 256, 0, stream>>>(scb, CL);

    // attn = P @ V : M=L, N=DH, K=L
    gemm(stream, scb, vtb, attnb, nullptr, nullptr, nullptr,
         CL, CDH, CL, CL, CL, CD, 0, 0, 1,
         CB * CH, CH,
         (long long)CH * CL * CL, (long long)CL * CL,
         (long long)CH * CDH * CL, (long long)CDH * CL,
         (long long)CL * CD, CDH);

    // h = h + attn @ Wo^T  (fp32 out + residual)
    gemm(stream, attnb, Wo, h, nullptr, h, nullptr,
         ROWS, CD, CD, CD, CD, CD, 0, 0, 0, 1, 1, 0,0,0,0,0,0);

    layernorm_kernel<<<ROWS, 256, 0, stream>>>(h, ln2_g + i * CD, ln2_b + i * CD, xnb, CD);

    // ff = gelu(xn @ W1^T + b1)  (bf16 out)
    gemm(stream, xnb, W1, ffab, ff_b1 + (size_t)i * CDFF, nullptr, nullptr,
         ROWS, CDFF, CD, CD, CD, CDFF, 1, 0, 1, 1, 1, 0,0,0,0,0,0);

    // h = h + ff @ W2^T + b2  (fp32 out + residual)
    gemm(stream, ffab, W2, h, ff_b2 + (size_t)i * CD, h, nullptr,
         ROWS, CD, CDFF, CDFF, CDFF, CD, 0, 0, 0, 1, 1, 0,0,0,0,0,0);
  }

  // final LN + tied LM head
  layernorm_kernel<<<ROWS, 256, 0, stream>>>(h, lnf_g, lnf_b, xnb, CD);
  gemm(stream, xnb, embb, logits, nullptr, nullptr, nullptr,
       ROWS, CVOCAB, CD, CD, CD, CVOCAB, 0, 0, 0, 1, 1, 0,0,0,0,0,0);
}
